// ScaledDotProductAttention2_549755814005
// MI455X (gfx1250) — matrix operand order repaired and checked by reading, not compiled, not run on hardware
//
#include <hip/hip_runtime.h>
#include <cmath>

typedef __attribute__((ext_vector_type(16))) __bf16 bf16x16;
typedef __attribute__((ext_vector_type(8)))  __bf16 bf16x8;
typedef __attribute__((ext_vector_type(4)))  __bf16 bf16x4;
typedef __attribute__((ext_vector_type(8)))  float  f32x8;
typedef __attribute__((ext_vector_type(4)))  int    i32x4;
typedef __attribute__((address_space(1))) i32x4* gbl_i32x4p;
typedef __attribute__((address_space(3))) i32x4* lds_i32x4p;

#define B_   8
#define LQ   2048
#define LK   2048
#define DD   640
#define INV_TEMP (1.0f / 13.0f)

#if __has_builtin(__builtin_amdgcn_global_load_async_to_lds_b128)
#define HAVE_ASYNC_LDS 1
#else
#define HAVE_ASYNC_LDS 0
#endif

// Dynamic LDS layout (bytes) -- ISA allows up to 320KB per workgroup:
//  [0,      131072) float  Ls[16][2048]   logits
//  [131072, 151552) __bf16 As[16][640]    q tile bf16 (GEMM1 A)
//  [151552, 217088) __bf16 Ab[16][2048]   attn probs bf16 (GEMM2 A)
//  [217088, 218112) float  red[256]
//  [218112, 218176) float  qs[16]
//  [218176, 218240) float  rowmax[16]
//  [218240, 218304) float  rowlse[16]
#define LDS_BYTES 218304

// ---------------------------------------------------------------------------
// Kernel 0: row squared-norms of q and k (one wave32 per row).
// ---------------------------------------------------------------------------
__global__ __launch_bounds__(256)
void rownorm_kernel(const float* __restrict__ q, const float* __restrict__ k,
                    float* __restrict__ qq, float* __restrict__ kk) {
  const int w    = blockIdx.x * (blockDim.x >> 5) + (threadIdx.x >> 5);
  const int lane = threadIdx.x & 31;
  const int nrows = B_ * LQ;
  const float* src; float* dst; int row;
  if (w < nrows) { src = q; dst = qq; row = w; }
  else           { src = k; dst = kk; row = w - nrows; }
  const float* p = src + (size_t)row * DD;
  float s = 0.0f;
  for (int d = lane; d < DD; d += 32) { const float x = p[d]; s += x * x; }
  #pragma unroll
  for (int off = 16; off > 0; off >>= 1) s += __shfl_xor(s, off, 32);
  if (lane == 0) dst[row] = s;
}

// ---------------------------------------------------------------------------
// Kernel 0b: elementwise f32 -> bf16 (for k, row-major preserved).
// ---------------------------------------------------------------------------
__global__ __launch_bounds__(256)
void cvt_bf16_kernel(const float* __restrict__ src, __bf16* __restrict__ dst,
                     int n4) {
  for (int i = blockIdx.x * 256 + threadIdx.x; i < n4; i += gridDim.x * 256) {
    const float4 f = ((const float4*)src)[i];
    bf16x4 r;
    r[0] = (__bf16)f.x; r[1] = (__bf16)f.y; r[2] = (__bf16)f.z; r[3] = (__bf16)f.w;
    *(bf16x4*)(dst + 4 * (size_t)i) = r;
  }
}

// ---------------------------------------------------------------------------
// Kernel 0c: v (B,LK,DD) f32 -> vt (B,DD,LK) bf16, LDS-tiled 32x32 transpose.
// ---------------------------------------------------------------------------
__global__ __launch_bounds__(256)
void cvt_transpose_v_kernel(const float* __restrict__ v, __bf16* __restrict__ vt) {
  __shared__ float tile[32][33];
  const int ncol = DD / 32;                 // 20 column tiles
  const int per_b = (LK / 32) * ncol;       // 64*20 tiles per batch
  const int b   = blockIdx.x / per_b;
  const int rem = blockIdx.x % per_b;
  const int k0  = (rem / ncol) * 32;
  const int n0  = (rem % ncol) * 32;
  const int tx = threadIdx.x & 31, ty = threadIdx.x >> 5;  // 32 x 8
  const float* vb = v + (size_t)b * LK * DD;
  #pragma unroll
  for (int s = 0; s < 4; ++s)
    tile[ty + s * 8][tx] = vb[(size_t)(k0 + ty + s * 8) * DD + n0 + tx];
  __syncthreads();
  __bf16* vtb = vt + (size_t)b * DD * LK;
  #pragma unroll
  for (int s = 0; s < 4; ++s)
    vtb[(size_t)(n0 + ty + s * 8) * LK + k0 + tx] = (__bf16)tile[tx][ty + s * 8];
}

// ---------------------------------------------------------------------------
// Kernel 1: fused L2-attention for one (batch, 16-query-row) tile.
// PRECVT: kb16 = bf16 k (row-major), vt16 = bf16 v transposed (B,DD,LK);
//         B-fragments become single contiguous 32B loads, no inner-loop cvt.
// ---------------------------------------------------------------------------
template <bool PRECVT>
__global__ __launch_bounds__(256)
void l2attn_kernel(const float* __restrict__ q, const float* __restrict__ k,
                   const float* __restrict__ v,
                   const __bf16* __restrict__ kb16, const __bf16* __restrict__ vt16,
                   const float* __restrict__ qqg, const float* __restrict__ kkg,
                   float* __restrict__ outp, float* __restrict__ attnp,
                   float* __restrict__ lattnp) {
  extern __shared__ char smem[];
  float*  Ls     = (float*)smem;
  __bf16* As     = (__bf16*)(smem + 131072);
  __bf16* Ab     = (__bf16*)(smem + 151552);
  float*  red    = (float*)(smem + 217088);
  float*  qs     = (float*)(smem + 218112);
  float*  rowmax = (float*)(smem + 218176);
  float*  rowlse = (float*)(smem + 218240);

  const int tid  = threadIdx.x;
  const int lane = tid & 31;
  const int wave = tid >> 5;
  const int b    = blockIdx.x >> 7;
  const int m0   = (blockIdx.x & 127) << 4;

  const float* qb = q + ((size_t)b * LQ + m0) * DD;
  const float* kb = k + (size_t)b * LK * DD;
  const float* vb = v + (size_t)b * LK * DD;

  // ---- Stage q tile (async global->LDS when available), convert to bf16 --
#if HAVE_ASYNC_LDS
  {
    const char* gq = (const char*)qb;
    char* lq = (char*)Ls;                 // Ls is free at this point
    #pragma unroll
    for (int it = 0; it < 10; ++it) {     // 16*640*4 B = 2560 x 16B
      const int off = (tid + it * 256) * 16;
      __builtin_amdgcn_global_load_async_to_lds_b128(
          (gbl_i32x4p)(gq + off), (lds_i32x4p)(lq + off), 0, 0);
    }
  }
#if __has_builtin(__builtin_amdgcn_s_wait_asynccnt)
  __builtin_amdgcn_s_wait_asynccnt(0);
#else
  asm volatile("s_wait_asynccnt 0" ::: "memory");
#endif
  __syncthreads();
  for (int idx = tid; idx < 16 * DD; idx += 256) As[idx] = (__bf16)Ls[idx];
#else
  for (int idx = tid; idx < 16 * DD; idx += 256) As[idx] = (__bf16)qb[idx];
#endif
  if (tid < 16) qs[tid] = qqg[b * LQ + m0 + tid];
  __syncthreads();

  const int lhalf = lane >> 4;
  const int lmod  = lane & 15;

  // ---- GEMM1: logits = (2 q.k - ||q||^2 - ||k||^2)/13, N=2048 ------------
  // 8 waves x 4 groups x 4 subtiles; A-fragment shared by 4 WMMAs.
  for (int g = 0; g < 4; ++g) {
    const int nb = wave * 256 + g * 64;
    f32x8 acc[4] = {};
    for (int kc = 0; kc < DD; kc += 32) {
      bf16x16 af;
      {
        const bf16x8 a0 = *(const bf16x8*)&As[lmod * DD + kc + (lhalf << 4)];
        const bf16x8 a1 = *(const bf16x8*)&As[lmod * DD + kc + (lhalf << 4) + 8];
        #pragma unroll
        for (int i = 0; i < 8; ++i) { af[i] = a0[i]; af[8 + i] = a1[i]; }
      }
      #pragma unroll
      for (int u = 0; u < 4; ++u) {
        const int n = nb + u * 16 + lmod;
        bf16x16 bfv;
        if (PRECVT) {
          const __bf16* kp = kb16 + (size_t)n * DD + kc + (lhalf << 4);
          __builtin_prefetch(kp + 32, 0, 1);
          bfv = *(const bf16x16*)kp;
        } else {
          const float* kp = kb + (size_t)n * DD + kc + (lhalf << 4);
          __builtin_prefetch(kp + 32, 0, 1);
          #pragma unroll
          for (int j = 0; j < 8; ++j) {
            const float2 f = *(const float2*)(kp + 2 * j);
            bfv[2 * j]     = (__bf16)f.x;
            bfv[2 * j + 1] = (__bf16)f.y;
          }
        }
        acc[u] = __builtin_amdgcn_wmma_f32_16x16x32_bf16(
            false, af, false, bfv, (short)0, acc[u], false, false);
      }
    }
    #pragma unroll
    for (int u = 0; u < 4; ++u) {
      const int n = nb + u * 16 + lmod;
      const float kkv = kkg[b * LK + n];
      #pragma unroll
      for (int r = 0; r < 8; ++r) {
        const int m = r + (lhalf << 3);
        Ls[m * LK + n] = (2.0f * acc[u][r] - qs[m] - kkv) * INV_TEMP;
      }
    }
  }
  __syncthreads();

  // ---- Row-wise log-softmax statistics (16 threads per row) --------------
  {
    const int r = tid >> 4, s = tid & 15;
    float mx = -INFINITY;
    for (int c = s; c < LK; c += 16) mx = fmaxf(mx, Ls[r * LK + c]);
    red[tid] = mx;
  }
  __syncthreads();
  if (tid < 16) {
    float mx = red[tid * 16];
    #pragma unroll
    for (int j = 1; j < 16; ++j) mx = fmaxf(mx, red[tid * 16 + j]);
    rowmax[tid] = mx;
  }
  __syncthreads();
  {
    const int r = tid >> 4, s = tid & 15;
    const float mx = rowmax[r];
    float sm = 0.0f;
    for (int c = s; c < LK; c += 16) sm += __expf(Ls[r * LK + c] - mx);
    red[tid] = sm;
  }
  __syncthreads();
  if (tid < 16) {
    float sm = 0.0f;
    #pragma unroll
    for (int j = 0; j < 16; ++j) sm += red[tid * 16 + j];
    rowlse[tid] = rowmax[tid] + __logf(sm);
  }
  __syncthreads();

  // ---- Emit attn & log_attn; stash probs as bf16 for GEMM2 ---------------
  float* attn_out = attnp  + ((size_t)b * LQ + m0) * LK;
  float* lat_out  = lattnp + ((size_t)b * LQ + m0) * LK;
  for (int idx = tid; idx < 16 * LK; idx += 256) {
    const int r = idx >> 11;
    const float la = Ls[idx] - rowlse[r];
    const float pr = __expf(la);
    lat_out[idx]  = la;
    attn_out[idx] = pr;
    Ab[idx]       = (__bf16)pr;
  }
  __syncthreads();

  // ---- GEMM2: out[16][640] = attn[16][2048] x v[2048][640] ---------------
  // 8 waves x 5 subtiles, k-outer: one A-fragment feeds 5 WMMAs.
  float* out_tile = outp + ((size_t)b * LQ + m0) * DD;
  const __bf16* vtb = PRECVT ? (vt16 + (size_t)b * DD * LK) : nullptr;
  {
    f32x8 acc[5] = {};
    for (int kc = 0; kc < LK; kc += 32) {
      bf16x16 af;
      {
        const bf16x8 a0 = *(const bf16x8*)&Ab[lmod * LK + kc + (lhalf << 4)];
        const bf16x8 a1 = *(const bf16x8*)&Ab[lmod * LK + kc + (lhalf << 4) + 8];
        #pragma unroll
        for (int i = 0; i < 8; ++i) { af[i] = a0[i]; af[8 + i] = a1[i]; }
      }
      #pragma unroll
      for (int t = 0; t < 5; ++t) {
        const int n = ((wave * 5 + t) << 4) + lmod;
        bf16x16 bfv;
        if (PRECVT) {
          const __bf16* vp = vtb + (size_t)n * LK + kc + (lhalf << 4);
          __builtin_prefetch(vp + 32, 0, 1);
          bfv = *(const bf16x16*)vp;
        } else {
          const int kb0 = kc + (lhalf << 4);
          #pragma unroll
          for (int j = 0; j < 8; ++j) {
            bfv[2 * j]     = (__bf16)vb[(size_t)(kb0 + 2 * j)     * DD + n];
            bfv[2 * j + 1] = (__bf16)vb[(size_t)(kb0 + 2 * j + 1) * DD + n];
          }
        }
        acc[t] = __builtin_amdgcn_wmma_f32_16x16x32_bf16(
            false, af, false, bfv, (short)0, acc[t], false, false);
      }
    }
    #pragma unroll
    for (int t = 0; t < 5; ++t) {
      const int n0 = (wave * 5 + t) << 4;
      #pragma unroll
      for (int r = 0; r < 8; ++r) {
        const int m = r + (lhalf << 3);
        out_tile[m * DD + n0 + lmod] = acc[t][r];
      }
    }
  }
}

// ---------------------------------------------------------------------------
extern "C" void kernel_launch(void* const* d_in, const int* in_sizes, int n_in,
                              void* d_out, int out_size, void* d_ws, size_t ws_size,
                              hipStream_t stream) {
  const float* q = (const float*)d_in[0];
  const float* k = (const float*)d_in[1];
  const float* v = (const float*)d_in[2];

  float* out   = (float*)d_out;                       // (B, Lq, D)
  float* attn  = out  + (size_t)B_ * LQ * DD;         // (B, Lq, Lk)
  float* lattn = attn + (size_t)B_ * LQ * LK;         // (B, Lq, Lk)

  const size_t qqkk_bytes = (size_t)2 * B_ * LQ * sizeof(float);
  const size_t cvt_bytes  = (size_t)B_ * LK * DD * sizeof(__bf16);  // ~21 MB each
  float* qq = (float*)d_ws;
  float* kk = qq + (size_t)B_ * LQ;
  __bf16* kb16 = (__bf16*)((char*)d_ws + qqkk_bytes);
  __bf16* vt16 = kb16 + (size_t)B_ * LK * DD;
  const bool precvt = ws_size >= qqkk_bytes + 2 * cvt_bytes;

  rownorm_kernel<<<(2 * B_ * LQ) / 8, 256, 0, stream>>>(q, k, qq, kk);
  if (precvt) {
    cvt_bf16_kernel<<<4096, 256, 0, stream>>>(k, kb16, (B_ * LK * DD) / 4);
    cvt_transpose_v_kernel<<<B_ * (LK / 32) * (DD / 32), 256, 0, stream>>>(v, vt16);
    l2attn_kernel<true><<<B_ * (LQ / 16), 256, LDS_BYTES, stream>>>(
        q, k, v, kb16, vt16, qq, kk, out, attn, lattn);
  } else {
    l2attn_kernel<false><<<B_ * (LQ / 16), 256, LDS_BYTES, stream>>>(
        q, k, v, nullptr, nullptr, qq, kk, out, attn, lattn);
  }
}
